// Mamba2UniQLMixer_83416854823018
// MI455X (gfx1250) — matrix operand order repaired and checked by reading, not compiled, not run on hardware
//
#include <hip/hip_runtime.h>
#include <hip/hip_bf16.h>

typedef __attribute__((ext_vector_type(16))) _Float16 v16h;
typedef __attribute__((ext_vector_type(8)))  _Float16 v8h;
typedef __attribute__((ext_vector_type(8)))  float    v8f;

static constexpr int cB = 2, cL = 4096, cD = 2048, cH = 32, cP = 64, cN = 128;
static constexpr int cINTER = 2048, cCONV = 2304, cPROJ = 4384;
static constexpr int cPROJP = 4480;             // PROJ padded to multiple of 128
static constexpr int cM = cB * cL;              // 8192 rows for both GEMMs

// ---------------------------------------------------------------------------
// fp32 -> f16 conversion
// ---------------------------------------------------------------------------
__global__ __launch_bounds__(256) void f32_to_f16_kernel(
    const float* __restrict__ in, _Float16* __restrict__ out, size_t n)
{
    size_t i = (size_t)blockIdx.x * 256 + threadIdx.x;
    if (i < n) out[i] = (_Float16)in[i];
}

// W_in (PROJ x D) -> f16 padded to (PROJP x D); pad rows are zero so the
// GEMM needs no edge guards (keeps EXEC all-ones around every WMMA).
__global__ __launch_bounds__(256) void win_to_f16_pad_kernel(
    const float* __restrict__ in, _Float16* __restrict__ out)
{
    size_t i = (size_t)blockIdx.x * 256 + threadIdx.x;   // over PROJP*D
    if (i >= (size_t)cPROJP * cD) return;
    size_t r = i >> 11;                                  // D = 2048
    out[i] = (r < (size_t)cPROJ) ? (_Float16)in[i] : (_Float16)0.f;
}

// ---------------------------------------------------------------------------
// GEMM: C[M,N] = A[M,K] * Bt[N,K]^T   (A,Bt f16 row-major, C fp32 stride N)
// Requirements: M % 128 == 0, N % 128 == 0, K % 32 == 0 (pre-padded).
// Block: 256 threads = 8 waves; block tile 128(M) x 128(N).
// Wave computes 32x64: 2 A-frags x 4 B-frags -> 8 WMMAs per K-step,
// completely branch-free so EXEC is all-ones for every WMMA.
// ---------------------------------------------------------------------------
static __device__ __forceinline__ v16h load_afrag(const _Float16* p)
{
    // lane<16 holds K {0..7, 16..23}, lane>=16 holds {8..15, 24..31};
    // caller pre-offsets p by 8*kHalf.
    v8h lo = *(const v8h*)(p);
    v8h hi = *(const v8h*)(p + 16);
    v16h f;
#pragma unroll
    for (int i = 0; i < 8; ++i) { f[i] = lo[i]; f[i + 8] = hi[i]; }
    return f;
}

__global__ __launch_bounds__(256) void gemm_wmma_f16_kernel(
    const _Float16* __restrict__ A, const _Float16* __restrict__ Bt,
    float* __restrict__ C, int N, int K)
{
    const int lane  = threadIdx.x & 31;
    const int wave  = threadIdx.x >> 5;
    const int mBase = blockIdx.y * 128 + (wave & 3) * 32;   // 2 m-tiles of 16
    const int nBase = blockIdx.x * 128 + (wave >> 2) * 64;  // 4 n-tiles of 16
    const int l16   = lane & 15;
    const int kHalf = lane >> 4;

    const _Float16* aRow0 = A + (size_t)(mBase + l16) * K + 8 * kHalf;
    const _Float16* aRow1 = aRow0 + (size_t)16 * K;
    const _Float16* bCol  = Bt + (size_t)(nBase + l16) * K + 8 * kHalf;
    const size_t bTilePitch = (size_t)16 * K;               // 16 columns

    v8f acc[2][4] = {};

    for (int k0 = 0; k0 < K; k0 += 32) {
        // prefetch two K-steps ahead (speculative; OOB overshoot is dropped)
        __builtin_prefetch(aRow0 + k0 + 64, 0, 1);
        __builtin_prefetch(aRow1 + k0 + 64, 0, 1);
        __builtin_prefetch(bCol + k0 + 64, 0, 1);
        __builtin_prefetch(bCol + 3 * bTilePitch + k0 + 64, 0, 1);

        const v16h a0 = load_afrag(aRow0 + k0);
        const v16h a1 = load_afrag(aRow1 + k0);
#pragma unroll
        for (int t = 0; t < 4; ++t) {
            const v16h bf = load_afrag(bCol + (size_t)t * bTilePitch + k0);
            acc[0][t] = __builtin_amdgcn_wmma_f32_16x16x32_f16(
                false, a0, false, bf, (short)0, acc[0][t], false, false);
            acc[1][t] = __builtin_amdgcn_wmma_f32_16x16x32_f16(
                false, a1, false, bf, (short)0, acc[1][t], false, false);
        }
    }

    // D layout: VGPR r -> M = r + 8*(lane>>4), N = lane&15
    const int mOff = kHalf * 8;
#pragma unroll
    for (int mt = 0; mt < 2; ++mt) {
#pragma unroll
        for (int t = 0; t < 4; ++t) {
            float* cp = C + (size_t)(mBase + mt * 16 + mOff) * N
                          + nBase + t * 16 + l16;
#pragma unroll
            for (int r = 0; r < 8; ++r)
                cp[(size_t)r * N] = acc[mt][t][r];
        }
    }
}

// ---------------------------------------------------------------------------
// Depthwise causal conv (K=4) + bias + SiLU on the hBC slice of proj
// ---------------------------------------------------------------------------
__global__ __launch_bounds__(256) void conv_silu_kernel(
    const float* __restrict__ proj, const float* __restrict__ cw,
    const float* __restrict__ cb, float* __restrict__ hbc)
{
    size_t i = (size_t)blockIdx.x * 256 + threadIdx.x;   // over B*L*CONV
    if (i >= (size_t)cB * cL * cCONV) return;
    const int c     = (int)(i % cCONV);
    const size_t bl = i / cCONV;
    const int l     = (int)(bl % cL);
    const size_t b0 = bl - l;                            // b*L
    float acc = cb[c];
#pragma unroll
    for (int k = 0; k < 4; ++k) {
        const int ll = l - 3 + k;
        if (ll >= 0)
            acc += cw[c * 4 + k] *
                   proj[(b0 + ll) * (size_t)cPROJP + cINTER + c];
    }
    hbc[i] = acc / (1.f + __expf(-acc));                 // SiLU
}

// ---------------------------------------------------------------------------
// dt = softplus(dt_raw + bias); dA = exp(dt * (-exp(A_log)))
// ---------------------------------------------------------------------------
__global__ __launch_bounds__(256) void dt_prep_kernel(
    const float* __restrict__ proj, const float* __restrict__ dt_bias,
    const float* __restrict__ A_log, float* __restrict__ dt,
    float* __restrict__ dA)
{
    const int i = blockIdx.x * 256 + threadIdx.x;        // over B*L*H
    if (i >= cB * cL * cH) return;
    const int h  = i & (cH - 1);
    const int bl = i >> 5;
    const float x  = proj[(size_t)bl * cPROJP + cINTER + cCONV + h] + dt_bias[h];
    const float sp = (x > 20.f) ? x : log1pf(__expf(x));
    dt[i] = sp;
    dA[i] = __expf(sp * (-__expf(A_log[h])));
}

// ---------------------------------------------------------------------------
// Selective scan: one block per (b,h). State h[P=64][N=128] in registers:
// thread t owns p = t>>2, n-segment = (t&3)*32 (32 floats each).
// B_t/C_t/x_t staged through LDS each step; y reduced over the 4 lanes
// sharing p with wave32 shuffles.
// ---------------------------------------------------------------------------
__global__ __launch_bounds__(256) void scan_kernel(
    const float* __restrict__ hbc, const float* __restrict__ dt,
    const float* __restrict__ dA, const float* __restrict__ Dp,
    float* __restrict__ y)
{
    const int bh = blockIdx.x;           // 0..63
    const int b  = bh >> 5;
    const int h  = bh & 31;
    const int t  = threadIdx.x;
    const int p  = t >> 2;               // 0..63
    const int ns = (t & 3) * 32;         // n segment base

    __shared__ float sB[cN], sC[cN], sX[cP], sDt, sDa;

    float st[32];
#pragma unroll
    for (int j = 0; j < 32; ++j) st[j] = 0.f;
    const float Dh = Dp[h];

    for (int l = 0; l < cL; ++l) {
        const size_t row  = (size_t)b * cL + l;
        const size_t hOff = row * cCONV;
        if (t < 128)  sB[t]       = hbc[hOff + cINTER + t];
        else          sC[t - 128] = hbc[hOff + cINTER + cN + (t - 128)];
        if (t < 64)   sX[t]       = hbc[hOff + h * cP + t];
        if (t == 254) sDt         = dt[row * cH + h];
        if (t == 255) sDa         = dA[row * cH + h];
        __syncthreads();

        const float xv = sX[p];
        const float u  = sDt * xv;
        const float da = sDa;
        float acc = 0.f;
#pragma unroll
        for (int j = 0; j < 32; ++j) {
            const int n = ns + j;
            const float s = da * st[j] + u * sB[n];
            st[j] = s;
            acc += s * sC[n];
        }
        // 4 consecutive lanes share p -> butterfly reduce within wave32
        acc += __shfl_xor(acc, 1, 32);
        acc += __shfl_xor(acc, 2, 32);
        if ((t & 3) == 0)
            y[(row * cH + h) * cP + p] = acc + Dh * xv;
        __syncthreads();
    }
}

// ---------------------------------------------------------------------------
// g = y * silu(gate); RMS-norm over INTER; emit f16 for the out-GEMM
// ---------------------------------------------------------------------------
__global__ __launch_bounds__(256) void gatenorm_kernel(
    const float* __restrict__ y, const float* __restrict__ proj,
    const float* __restrict__ normw, _Float16* __restrict__ g)
{
    const int row = blockIdx.x;          // b*L + l
    const int t   = threadIdx.x;
    const float* yr = y    + (size_t)row * cINTER;
    const float* gr = proj + (size_t)row * cPROJP;  // gate = first INTER cols

    __shared__ float red[8];
    __shared__ float sScale;
    float vals[8];
    float ss = 0.f;
#pragma unroll
    for (int i = 0; i < 8; ++i) {
        const int idx = t + i * 256;
        const float gt = gr[idx];
        const float v  = yr[idx] * (gt / (1.f + __expf(-gt)));
        vals[i] = v;
        ss += v * v;
    }
#pragma unroll
    for (int off = 16; off > 0; off >>= 1) ss += __shfl_xor(ss, off, 32);
    if ((t & 31) == 0) red[t >> 5] = ss;
    __syncthreads();
    if (t == 0) {
        float s = 0.f;
        for (int i = 0; i < 8; ++i) s += red[i];
        sScale = rsqrtf(s / (float)cINTER + 1e-5f);
    }
    __syncthreads();
    const float sc = sScale;
#pragma unroll
    for (int i = 0; i < 8; ++i) {
        const int idx = t + i * 256;
        g[(size_t)row * cINTER + idx] = (_Float16)(vals[i] * sc * normw[idx]);
    }
}

// ---------------------------------------------------------------------------
extern "C" void kernel_launch(void* const* d_in, const int* in_sizes, int n_in,
                              void* d_out, int out_size, void* d_ws, size_t ws_size,
                              hipStream_t stream)
{
    const float* hidden  = (const float*)d_in[0];
    const float* W_in    = (const float*)d_in[1];
    const float* conv_w  = (const float*)d_in[2];
    const float* conv_b  = (const float*)d_in[3];
    const float* dt_bias = (const float*)d_in[4];
    const float* A_log   = (const float*)d_in[5];
    const float* D_param = (const float*)d_in[6];
    const float* norm_w  = (const float*)d_in[7];
    const float* W_out   = (const float*)d_in[8];
    float* out = (float*)d_out;

    char* ws = (char*)d_ws;
    size_t off = 0;
    auto alloc = [&](size_t bytes) {
        size_t p = off;
        off = (off + bytes + 255) & ~(size_t)255;
        return p;
    };
    _Float16* hxf16 = (_Float16*)(ws + alloc((size_t)cM * cD * 2));
    _Float16* wif16 = (_Float16*)(ws + alloc((size_t)cPROJP * cD * 2));
    _Float16* wof16 = (_Float16*)(ws + alloc((size_t)cD * cINTER * 2));
    float*    proj  = (float*)   (ws + alloc((size_t)cM * cPROJP * 4));
    float*    hbc   = (float*)   (ws + alloc((size_t)cM * cCONV * 4));
    float*    dtb   = (float*)   (ws + alloc((size_t)cM * cH * 4));
    float*    dab   = (float*)   (ws + alloc((size_t)cM * cH * 4));
    float*    ybuf  = (float*)   (ws + alloc((size_t)cM * cINTER * 4));
    _Float16* g16   = (_Float16*)(ws + alloc((size_t)cM * cINTER * 2));

    // 1) one-time fp32 -> f16 conversions (W_in zero-padded to 4480 rows)
    {
        size_t n = (size_t)cM * cD;
        f32_to_f16_kernel<<<(unsigned)((n + 255) / 256), 256, 0, stream>>>(hidden, hxf16, n);
        n = (size_t)cPROJP * cD;
        win_to_f16_pad_kernel<<<(unsigned)((n + 255) / 256), 256, 0, stream>>>(W_in, wif16);
        n = (size_t)cD * cINTER;
        f32_to_f16_kernel<<<(unsigned)((n + 255) / 256), 256, 0, stream>>>(W_out, wof16, n);
    }

    // 2) in-projection GEMM: proj[M, PROJP] (branch-free, padded N)
    {
        dim3 grid(cPROJP / 128, cM / 128);
        gemm_wmma_f16_kernel<<<grid, 256, 0, stream>>>(hxf16, wif16, proj, cPROJP, cD);
    }

    // 3) depthwise causal conv + SiLU
    {
        size_t n = (size_t)cM * cCONV;
        conv_silu_kernel<<<(unsigned)((n + 255) / 256), 256, 0, stream>>>(proj, conv_w, conv_b, hbc);
    }

    // 4) dt / dA prep
    dt_prep_kernel<<<(cM * cH) / 256, 256, 0, stream>>>(proj, dt_bias, A_log, dtb, dab);

    // 5) selective scan (one block per (b,h))
    scan_kernel<<<cB * cH, 256, 0, stream>>>(hbc, dtb, dab, D_param, ybuf);

    // 6) gate * SiLU + RMS norm -> f16
    gatenorm_kernel<<<cM, 256, 0, stream>>>(ybuf, proj, norm_w, g16);

    // 7) out-projection GEMM: out[M, D]
    {
        dim3 grid(cD / 128, cM / 128);
        gemm_wmma_f16_kernel<<<grid, 256, 0, stream>>>(g16, wof16, out, cD, cINTER);
    }
}
